// MultiHeadSelfAttention_25245817766315
// MI455X (gfx1250) — matrix operand order repaired and checked by reading, not compiled, not run on hardware
//
#include <hip/hip_runtime.h>

#define N_TOK 2048
#define DIM 1024
#define NH 16
#define DK 64

typedef __attribute__((ext_vector_type(16))) _Float16 v16h;
typedef __attribute__((ext_vector_type(8)))  _Float16 v8h;
typedef __attribute__((ext_vector_type(8)))  float    v8f;

union HFrag {
    v16h v;
    v8h  h[2];
};

// ---------------------------------------------------------------------------
// Kernel 1: f32 -> f16 conversion, 8 elements per thread (vectorized)
// ---------------------------------------------------------------------------
__global__ __launch_bounds__(256)
void f32_to_f16_kernel(const float* __restrict__ src, _Float16* __restrict__ dst, int n) {
    int i = (blockIdx.x * blockDim.x + threadIdx.x) * 8;
    if (i + 7 < n) {
        float4 a = *(const float4*)(src + i);
        float4 b = *(const float4*)(src + i + 4);
        v8h o;
        o[0] = (_Float16)a.x; o[1] = (_Float16)a.y;
        o[2] = (_Float16)a.z; o[3] = (_Float16)a.w;
        o[4] = (_Float16)b.x; o[5] = (_Float16)b.y;
        o[6] = (_Float16)b.z; o[7] = (_Float16)b.w;
        *(v8h*)(dst + i) = o;
    }
}

// ---------------------------------------------------------------------------
// Kernel 2: Y = X @ W^T   (C[n,o] = sum_d X[n,d] * W[o,d]), f16 in, f16 out
//   X: [N_TOK, DIM] row-major, W: [DIM, DIM] row-major (torch Linear weight)
//   transposed==0 : Y[n*DIM + o]   (row-major, used for Q and K)
//   transposed==1 : Y[o*N_TOK + n] (column-major, used for V -> Vt)
// Workgroup: 256 threads = 8 waves, 128x128 output tile; wave tile 64x32.
// ---------------------------------------------------------------------------
__global__ __launch_bounds__(256)
void gemm_xwT_f16(const _Float16* __restrict__ X,
                  const _Float16* __restrict__ W,
                  _Float16* __restrict__ Y,
                  int transposed)
{
    const int lane = threadIdx.x & 31;
    const int wave = threadIdx.x >> 5;
    const int wr   = wave >> 2;                 // 0..1
    const int wc   = wave & 3;                  // 0..3
    const int row0 = blockIdx.x * 128 + wr * 64;
    const int col0 = blockIdx.y * 128 + wc * 32;

    const int lm  = lane & 15;
    const int hi  = lane >> 4;
    const int akb = hi * 8;    // A-frag k interleave base
    const int bkb = hi * 16;   // unused; B-frags also use akb

    v8f acc[4][2];
    #pragma unroll
    for (int i = 0; i < 4; ++i)
        #pragma unroll
        for (int j = 0; j < 2; ++j) acc[i][j] = v8f{};

    for (int kk = 0; kk < DIM; kk += 32) {
        HFrag a[4], b[2];
        #pragma unroll
        for (int ms = 0; ms < 4; ++ms) {
            const _Float16* p = X + (size_t)(row0 + ms * 16 + lm) * DIM + kk + akb;
            a[ms].h[0] = *(const v8h*)(p);        // K = akb+0..7
            a[ms].h[1] = *(const v8h*)(p + 16);   // K = 16+akb .. 16+akb+7
        }
        #pragma unroll
        for (int ns = 0; ns < 2; ++ns) {
            const _Float16* p = W + (size_t)(col0 + ns * 16 + lm) * DIM + kk + akb;
            b[ns].h[0] = *(const v8h*)(p);        // K = akb+0..7
            b[ns].h[1] = *(const v8h*)(p + 16);   // K = 16+akb .. 16+akb+7
        }
        #pragma unroll
        for (int ms = 0; ms < 4; ++ms)
            #pragma unroll
            for (int ns = 0; ns < 2; ++ns)
                acc[ms][ns] = __builtin_amdgcn_wmma_f32_16x16x32_f16(
                    false, a[ms].v, false, b[ns].v, (short)0, acc[ms][ns],
                    false, false);
    }

    #pragma unroll
    for (int ms = 0; ms < 4; ++ms) {
        #pragma unroll
        for (int ns = 0; ns < 2; ++ns) {
            const int o     = col0 + ns * 16 + lm;
            const int mbase = row0 + ms * 16 + hi * 8;
            if (!transposed) {
                #pragma unroll
                for (int j = 0; j < 8; ++j)
                    Y[(size_t)(mbase + j) * DIM + o] = (_Float16)acc[ms][ns][j];
            } else {
                v8h pk;
                #pragma unroll
                for (int j = 0; j < 8; ++j) pk[j] = (_Float16)acc[ms][ns][j];
                *(v8h*)(Y + (size_t)o * N_TOK + mbase) = pk;   // 8 rows contiguous
            }
        }
    }
}

// ---------------------------------------------------------------------------
// Kernel 3: flash attention. One wave handles (head h, 16-query block).
//   Q,K : [N_TOK, DIM] f16 row-major; Vt : [DIM, N_TOK] f16 (head-major cols)
//   out : [N_TOK, DIM] f32
// ---------------------------------------------------------------------------
__global__ __launch_bounds__(256)
void attn_kernel(const _Float16* __restrict__ Q,
                 const _Float16* __restrict__ K,
                 const _Float16* __restrict__ Vt,
                 float* __restrict__ out)
{
    __shared__ _Float16 lds[8][16 * 40];   // per-wave 16x32 P tile, padded stride

    const int lane = threadIdx.x & 31;
    const int wave = threadIdx.x >> 5;
    const int gw   = blockIdx.x * 8 + wave;
    const int h    = gw >> 7;            // / (N_TOK/16)
    const int q0   = (gw & 127) * 16;
    const int hoff = h * DK;

    const int lm  = lane & 15;
    const int hi  = lane >> 4;
    const int akb = hi * 8;
    const int bkb = hi * 16;

    _Float16* plds = &lds[wave][0];

    // Q A-fragments for d = 0..31 and 32..63 (kept in registers all loop)
    HFrag qf[2];
    #pragma unroll
    for (int f = 0; f < 2; ++f) {
        const _Float16* p = Q + (size_t)(q0 + lm) * DIM + hoff + 32 * f + akb;
        qf[f].h[0] = *(const v8h*)(p);
        qf[f].h[1] = *(const v8h*)(p + 16);
    }

    v8f o_acc[4];
    #pragma unroll
    for (int g = 0; g < 4; ++g) o_acc[g] = v8f{};
    float m_run[8], l_run[8];
    #pragma unroll
    for (int j = 0; j < 8; ++j) { m_run[j] = -1e30f; l_run[j] = 0.0f; }

    const float scale = 0.125f;   // 1/sqrt(64)

    for (int kb = 0; kb < N_TOK; kb += 32) {
        // ---- S = scale * Q K^T for a 16x32 tile (2 column tiles) ----
        v8f s[2];
        #pragma unroll
        for (int t = 0; t < 2; ++t) {
            HFrag kf[2];
            const int key = kb + t * 16 + lm;
            #pragma unroll
            for (int f = 0; f < 2; ++f) {
                const _Float16* p = K + (size_t)key * DIM + hoff + 32 * f + akb;
                kf[f].h[0] = *(const v8h*)(p);
                kf[f].h[1] = *(const v8h*)(p + 16);
            }
            v8f sv = v8f{};
            sv = __builtin_amdgcn_wmma_f32_16x16x32_f16(
                false, qf[0].v, false, kf[0].v, (short)0, sv, false, false);
            sv = __builtin_amdgcn_wmma_f32_16x16x32_f16(
                false, qf[1].v, false, kf[1].v, (short)0, sv, false, false);
            #pragma unroll
            for (int j = 0; j < 8; ++j) s[t][j] = sv[j] * scale;
        }

        // ---- online softmax; each row lives in one half-wave (width=16) ----
        #pragma unroll
        for (int j = 0; j < 8; ++j) {
            float cm = fmaxf(s[0][j], s[1][j]);
            cm = fmaxf(cm, __shfl_xor(cm, 1, 16));
            cm = fmaxf(cm, __shfl_xor(cm, 2, 16));
            cm = fmaxf(cm, __shfl_xor(cm, 4, 16));
            cm = fmaxf(cm, __shfl_xor(cm, 8, 16));
            const float mn = fmaxf(m_run[j], cm);
            const float sf = __expf(m_run[j] - mn);
            const float p0 = __expf(s[0][j] - mn);
            const float p1 = __expf(s[1][j] - mn);
            float rs = p0 + p1;
            rs += __shfl_xor(rs, 1, 16);
            rs += __shfl_xor(rs, 2, 16);
            rs += __shfl_xor(rs, 4, 16);
            rs += __shfl_xor(rs, 8, 16);
            l_run[j] = l_run[j] * sf + rs;
            m_run[j] = mn;
            #pragma unroll
            for (int g = 0; g < 4; ++g) o_acc[g][j] *= sf;
            // D-layout -> LDS (row = j + 8*hi, col = lm / lm+16)
            const int row = j + hi * 8;
            plds[row * 40 + lm]      = (_Float16)p0;
            plds[row * 40 + lm + 16] = (_Float16)p1;
        }
        asm volatile("s_wait_dscnt 0" ::: "memory");

        // ---- reload P as an A-fragment ----
        HFrag pf;
        {
            const _Float16* p = plds + lm * 40 + akb;
            pf.h[0] = *(const v8h*)(p);
            pf.h[1] = *(const v8h*)(p + 16);
        }

        // ---- O += P @ V ; V B-frags are contiguous thanks to Vt layout ----
        #pragma unroll
        for (int g = 0; g < 4; ++g) {
            HFrag vf;
            const _Float16* p = Vt + (size_t)(hoff + g * 16 + lm) * N_TOK + kb + akb;
            vf.h[0] = *(const v8h*)(p);
            vf.h[1] = *(const v8h*)(p + 16);
            o_acc[g] = __builtin_amdgcn_wmma_f32_16x16x32_f16(
                false, pf.v, false, vf.v, (short)0, o_acc[g], false, false);
        }
    }

    // ---- epilogue: divide by running sum, write f32 output ----
    #pragma unroll
    for (int g = 0; g < 4; ++g) {
        const int d = hoff + g * 16 + lm;
        #pragma unroll
        for (int j = 0; j < 8; ++j)
            out[(size_t)(q0 + j + hi * 8) * DIM + d] = o_acc[g][j] / l_run[j];
    }
}

// ---------------------------------------------------------------------------
// Launch
// ---------------------------------------------------------------------------
extern "C" void kernel_launch(void* const* d_in, const int* in_sizes, int n_in,
                              void* d_out, int out_size, void* d_ws, size_t ws_size,
                              hipStream_t stream) {
    const float* x  = (const float*)d_in[0];
    // d_in[1] = rela, unused by the module
    const float* Wq = (const float*)d_in[2];
    const float* Wk = (const float*)d_in[3];
    const float* Wv = (const float*)d_in[4];
    float* out = (float*)d_out;

    // workspace layout (f16): xh | wqh | wkh | wvh | qh | kh | vt   (22 MB)
    _Float16* xh  = (_Float16*)d_ws;
    _Float16* wqh = xh  + (size_t)N_TOK * DIM;
    _Float16* wkh = wqh + (size_t)DIM * DIM;
    _Float16* wvh = wkh + (size_t)DIM * DIM;
    _Float16* qh  = wvh + (size_t)DIM * DIM;
    _Float16* kh  = qh  + (size_t)N_TOK * DIM;
    _Float16* vt  = kh  + (size_t)N_TOK * DIM;

    const int nx = N_TOK * DIM;   // 2^21
    const int nw = DIM * DIM;     // 2^20
    f32_to_f16_kernel<<<nx / 8 / 256, 256, 0, stream>>>(x,  xh,  nx);
    f32_to_f16_kernel<<<nw / 8 / 256, 256, 0, stream>>>(Wq, wqh, nw);
    f32_to_f16_kernel<<<nw / 8 / 256, 256, 0, stream>>>(Wk, wkh, nw);
    f32_to_f16_kernel<<<nw / 8 / 256, 256, 0, stream>>>(Wv, wvh, nw);

    dim3 gg(N_TOK / 128, DIM / 128);   // (16, 8)
    gemm_xwT_f16<<<gg, 256, 0, stream>>>(xh, wqh, qh, 0);
    gemm_xwT_f16<<<gg, 256, 0, stream>>>(xh, wkh, kh, 0);
    gemm_xwT_f16<<<gg, 256, 0, stream>>>(xh, wvh, vt, 1);  // V stored transposed

    const int nwaves = NH * (N_TOK / 16);   // 2048 waves
    attn_kernel<<<nwaves / 8, 256, 0, stream>>>(qh, kh, vt, out);
}
